// MambaBlock_54279796687047
// MI455X (gfx1250) — compile-verified, matrix operand order checked by reading, not run on hardware
//
#include <hip/hip_runtime.h>

typedef __attribute__((ext_vector_type(16))) __bf16 v16bf;
typedef __attribute__((ext_vector_type(8)))  __bf16 v8bf;
typedef __attribute__((ext_vector_type(8)))  float  v8f;

#define NB    2
#define LSEQ  1024
#define DMD   1024
#define DI    2048
#define DS    16
#define DC    4
#define N_TOK (NB * LSEQ)          // 2048 tokens
#define NXP   (2 * DS + DI)        // 2080

__device__ __forceinline__ float sigmoidf_(float x) { return 1.f / (1.f + __expf(-x)); }
__device__ __forceinline__ float softplusf_(float x) { return x > 20.f ? x : log1pf(__expf(x)); }

// CDNA5 async global->LDS copy (ASYNCcnt-tracked, no VGPR data movement).
// lds_off: wave-relative LDS byte address (low 32 bits of a flat LDS pointer).
__device__ __forceinline__ void async_load_b128(unsigned lds_off, const void* gptr) {
  asm volatile("global_load_async_to_lds_b128 %0, %1, off"
               :: "v"(lds_off), "v"((unsigned long long)(size_t)gptr)
               : "memory");
}
__device__ __forceinline__ void wait_async_le4() {
  asm volatile("s_wait_asynccnt 4" ::: "memory");
}
__device__ __forceinline__ void wait_async_0() {
  asm volatile("s_wait_asynccnt 0" ::: "memory");
}

// ---------------------------------------------------------------------------
// f32 -> bf16 convert (weights)
// ---------------------------------------------------------------------------
__global__ __launch_bounds__(256)
void cvt_bf16(const float* __restrict__ s, __bf16* __restrict__ d, int n) {
  int i = blockIdx.x * 256 + threadIdx.x;
  if (i < n) d[i] = (__bf16)s[i];
}

// ---------------------------------------------------------------------------
// LayerNorm over DM=1024, one block per token, writes bf16 A-matrix for GEMM1
// ---------------------------------------------------------------------------
__global__ __launch_bounds__(256)
void ln_kernel(const float* __restrict__ x, const float* __restrict__ w,
               const float* __restrict__ b, __bf16* __restrict__ xn) {
  __shared__ float s1[256], s2[256];
  const int tok = blockIdx.x, tid = threadIdx.x;
  const float4 v = ((const float4*)(x + (size_t)tok * DMD))[tid];
  s1[tid] = v.x + v.y + v.z + v.w;
  s2[tid] = v.x * v.x + v.y * v.y + v.z * v.z + v.w * v.w;
  __syncthreads();
  for (int off = 128; off > 0; off >>= 1) {
    if (tid < off) { s1[tid] += s1[tid + off]; s2[tid] += s2[tid + off]; }
    __syncthreads();
  }
  const float mean = s1[0] * (1.f / DMD);
  const float var  = s2[0] * (1.f / DMD) - mean * mean;
  const float rstd = rsqrtf(var + 1e-5f);
  const int base = tid * 4;
  __bf16* o = xn + (size_t)tok * DMD + base;
  const float vv[4] = {v.x, v.y, v.z, v.w};
#pragma unroll
  for (int i = 0; i < 4; ++i)
    o[i] = (__bf16)((vv[i] - mean) * rstd * w[base + i] + b[base + i]);
}

// ---------------------------------------------------------------------------
// bf16 WMMA GEMM: C[M,N] = A[M,K] * W[K,N] (+ epilogue), f32 accumulate.
// Block tile 128x64x64, 256 threads = 8 waves (4 m-groups x 2 n-groups),
// each wave owns a 2x2 grid of 16x16 tiles -> 8 WMMA / K-block.
// A tile: async global->LDS (ping-pong), B tile: register double-buffer with
// transpose into LDS [n][k].
// epi: 0 = store; 1 = softplus(acc + bias[n]); 2 = acc + resid[m,n]
// ---------------------------------------------------------------------------
__global__ __launch_bounds__(256)
void gemm_bf16_wmma(const __bf16* __restrict__ A, const __bf16* __restrict__ W,
                    float* __restrict__ C, int M, int N, int K,
                    int epi, const float* __restrict__ bias,
                    const float* __restrict__ resid) {
  __shared__ __bf16 As[2][128][72];  // 144 B row stride -> 16B-aligned chunks
  __shared__ __bf16 Bs[64][72];      // transposed: Bs[n][k]

  const int tid  = threadIdx.x;
  const int lane = tid & 31;
  const int wave = tid >> 5;
  const int lrow = lane & 15, hf = lane >> 4;
  const int wm = wave >> 1;          // 0..3 : rows [wm*32, wm*32+32)
  const int wn = wave & 1;           // 0..1 : cols [wn*32, wn*32+32)
  const int bm0 = blockIdx.y * 128, bn0 = blockIdx.x * 64;
  const bool n_edge = (bn0 + 64) > N;
  const int nIter = K >> 6;

  v8f acc[2][2] = {{{}, {}}, {{}, {}}};

  // ---- staging helpers -----------------------------------------------------
  auto issueA = [&](int k0, int buf) {  // 128x64 bf16 = 1024 16B chunks, 4/thread
#pragma unroll
    for (int c = tid; c < 1024; c += 256) {
      const int row = c >> 3, col8 = (c & 7) << 3;
      async_load_b128((unsigned)(size_t)&As[buf][row][col8],
                      A + (size_t)(bm0 + row) * K + (k0 + col8));
    }
  };
  auto loadB = [&](int k0, v8bf* br) {  // 64x64 bf16 = 512 chunks, 2/thread
#pragma unroll
    for (int u = 0; u < 2; ++u) {
      const int c = tid + u * 256;
      const int kr = c >> 3, nc8 = (c & 7) << 3;
      const __bf16* src = W + (size_t)(k0 + kr) * N + (bn0 + nc8);
      if (!n_edge) {
        br[u] = *(const v8bf*)src;
      } else {
#pragma unroll
        for (int j = 0; j < 8; ++j)
          br[u][j] = (bn0 + nc8 + j) < N ? src[j] : (__bf16)0.f;
      }
    }
  };
  auto storeB = [&](const v8bf* br) {
#pragma unroll
    for (int u = 0; u < 2; ++u) {
      const int c = tid + u * 256;
      const int kr = c >> 3, nc8 = (c & 7) << 3;
#pragma unroll
      for (int j = 0; j < 8; ++j) Bs[nc8 + j][kr] = br[u][j];
    }
  };

  // ---- software pipeline ---------------------------------------------------
  issueA(0, 0);
  v8bf bcur[2], bnxt[2];
  loadB(0, bcur);

  for (int i = 0; i < nIter; ++i) {
    const int buf = i & 1;
    if (i) __syncthreads();            // everyone done reading Bs / As[buf^1]
    storeB(bcur);
    if (i + 1 < nIter) {
      issueA((i + 1) << 6, buf ^ 1);   // overlap with this iteration's WMMAs
      loadB((i + 1) << 6, bnxt);
      wait_async_le4();                // in-order: tile i's 4 asyncs retired
    } else {
      wait_async_0();
    }
    __syncthreads();                   // As[buf] + Bs visible to all waves

#pragma unroll
    for (int kk = 0; kk < 64; kk += 32) {
      v16bf af[2], bfrag[2];
#pragma unroll
      for (int mi = 0; mi < 2; ++mi) {
        const __bf16* ar = &As[buf][(wm * 2 + mi) * 16 + lrow][kk + hf * 8];
        af[mi] = __builtin_shufflevector(*(const v8bf*)ar, *(const v8bf*)(ar + 16),
                                         0, 1, 2, 3, 4, 5, 6, 7,
                                         8, 9, 10, 11, 12, 13, 14, 15);
      }
#pragma unroll
      for (int ni = 0; ni < 2; ++ni) {
        const __bf16* bp = &Bs[(wn * 2 + ni) * 16 + lrow][kk + hf * 8];
        bfrag[ni] = __builtin_shufflevector(*(const v8bf*)bp, *(const v8bf*)(bp + 16),
                                            0, 1, 2, 3, 4, 5, 6, 7,
                                            8, 9, 10, 11, 12, 13, 14, 15);
      }
#pragma unroll
      for (int mi = 0; mi < 2; ++mi)
#pragma unroll
        for (int ni = 0; ni < 2; ++ni)
          acc[mi][ni] = __builtin_amdgcn_wmma_f32_16x16x32_bf16(
              false, af[mi], false, bfrag[ni], (short)0, acc[mi][ni], false, false);
    }
    bcur[0] = bnxt[0];
    bcur[1] = bnxt[1];
  }

  // ---- epilogue + store: acc[v] -> row hf*8+v, col lane&15 ----
#pragma unroll
  for (int mi = 0; mi < 2; ++mi) {
#pragma unroll
    for (int ni = 0; ni < 2; ++ni) {
      const int gn = bn0 + (wn * 2 + ni) * 16 + lrow;
      if (gn >= N) continue;
#pragma unroll
      for (int v = 0; v < 8; ++v) {
        const int gm = bm0 + (wm * 2 + mi) * 16 + hf * 8 + v;
        if (gm >= M) continue;
        float val = acc[mi][ni][v];
        if (epi == 1)      val = softplusf_(val + bias[gn]);
        else if (epi == 2) val += resid[(size_t)gm * N + gn];
        C[(size_t)gm * N + gn] = val;
      }
    }
  }
}

// ---------------------------------------------------------------------------
// Depthwise causal conv (DC=4) + SiLU. Reads x_b = xz[..., :DI].
// ---------------------------------------------------------------------------
__global__ __launch_bounds__(256)
void conv_silu(const float* __restrict__ xz, const float* __restrict__ cw,
               const float* __restrict__ cb, float* __restrict__ xf,
               __bf16* __restrict__ xb) {
  const int idx = blockIdx.x * 256 + threadIdx.x;
  const int d = idx & (DI - 1);
  const int t = (idx >> 11) & (LSEQ - 1);
  const int bI = idx >> 21;
  float acc = cb[d];
#pragma unroll
  for (int j = 0; j < DC; ++j) {
    const int tt = t - (DC - 1) + j;
    if (tt >= 0)
      acc += cw[d * DC + j] * xz[((size_t)(bI * LSEQ + tt)) * (2 * DI) + d];
  }
  const float sv = acc * sigmoidf_(acc);
  xf[idx] = sv;
  xb[idx] = (__bf16)sv;
}

// delta_in = x_dbl[..., :DI] -> bf16 A-matrix for GEMM3
__global__ __launch_bounds__(256)
void din_cvt(const float* __restrict__ xdbl, __bf16* __restrict__ din) {
  const int idx = blockIdx.x * 256 + threadIdx.x;
  const int d = idx & (DI - 1);
  const size_t tok = (size_t)(idx >> 11);
  din[idx] = (__bf16)xdbl[tok * NXP + d];
}

// ---------------------------------------------------------------------------
// Selective scan: one thread per (b, d) channel, DS=16 state in registers.
// ---------------------------------------------------------------------------
__global__ __launch_bounds__(256)
void scan_kernel(const float* __restrict__ dlt, const float* __restrict__ xf,
                 const float* __restrict__ xdbl, const float* __restrict__ A_log,
                 float* __restrict__ y) {
  const int bI = blockIdx.y;
  const int d = blockIdx.x * 256 + threadIdx.x;
  float Av[DS];
#pragma unroll
  for (int n = 0; n < DS; ++n) Av[n] = -__expf(A_log[d * DS + n]);
  float h[DS];
#pragma unroll
  for (int n = 0; n < DS; ++n) h[n] = 0.f;

  for (int t = 0; t < LSEQ; ++t) {
    const size_t tok = (size_t)bI * LSEQ + t;
    if ((t & 15) == 0 && t + 16 < LSEQ) {       // global_prefetch_b8 ahead
      const size_t pt = (tok + 16) * DI + d;
      __builtin_prefetch(dlt + pt, 0, 1);
      __builtin_prefetch(xf + pt, 0, 1);
    }
    const float dl = dlt[tok * DI + d];
    const float xv = xf[tok * DI + d];
    const float du = dl * xv;
    const float* Bm = xdbl + tok * NXP + DI;
    const float* Cm = Bm + DS;
    float yv = 0.f;
#pragma unroll
    for (int n = 0; n < DS; ++n) {
      h[n] = __expf(dl * Av[n]) * h[n] + du * Bm[n];
      yv = fmaf(h[n], Cm[n], yv);
    }
    y[tok * DI + d] = yv;
  }
}

// y_total = (y_scan + x_b*D) * SiLU(z) -> bf16 A-matrix for GEMM4
__global__ __launch_bounds__(256)
void gate_kernel(const float* __restrict__ y, const float* __restrict__ xf,
                 const float* __restrict__ xz, const float* __restrict__ Dp,
                 __bf16* __restrict__ yg) {
  const int idx = blockIdx.x * 256 + threadIdx.x;
  const int d = idx & (DI - 1);
  const size_t tok = (size_t)(idx >> 11);
  const float yv = y[idx] + xf[idx] * Dp[d];
  const float zv = xz[tok * (size_t)(2 * DI) + DI + d];
  yg[idx] = (__bf16)(yv * (zv * sigmoidf_(zv)));
}

// ---------------------------------------------------------------------------
extern "C" void kernel_launch(void* const* d_in, const int* in_sizes, int n_in,
                              void* d_out, int out_size, void* d_ws, size_t ws_size,
                              hipStream_t stream) {
  (void)in_sizes; (void)n_in; (void)out_size; (void)ws_size;
  const float* x        = (const float*)d_in[0];
  const float* norm_w   = (const float*)d_in[1];
  const float* norm_b   = (const float*)d_in[2];
  const float* in_proj  = (const float*)d_in[3];
  const float* conv_w   = (const float*)d_in[4];
  const float* conv_b   = (const float*)d_in[5];
  const float* x_proj   = (const float*)d_in[6];
  const float* dt_w     = (const float*)d_in[7];
  const float* dt_b     = (const float*)d_in[8];
  const float* A_log    = (const float*)d_in[9];
  const float* D_param  = (const float*)d_in[10];
  const float* out_proj = (const float*)d_in[11];
  float* out = (float*)d_out;

  char* p = (char*)d_ws;
  auto alloc = [&](size_t bytes) -> void* {
    void* r = (void*)p;
    p += (bytes + 255) & ~(size_t)255;
    return r;
  };
  __bf16* w1b  = (__bf16*)alloc((size_t)DMD * 2 * DI * 2);
  __bf16* w2b  = (__bf16*)alloc((size_t)DI * NXP * 2);
  __bf16* w3b  = (__bf16*)alloc((size_t)DI * DI * 2);
  __bf16* w4b  = (__bf16*)alloc((size_t)DI * DMD * 2);
  __bf16* xnb  = (__bf16*)alloc((size_t)N_TOK * DMD * 2);
  float*  xz   = (float*) alloc((size_t)N_TOK * 2 * DI * 4);
  float*  xbsf = (float*) alloc((size_t)N_TOK * DI * 4);
  __bf16* xbsb = (__bf16*)alloc((size_t)N_TOK * DI * 2);
  float*  xdbl = (float*) alloc((size_t)N_TOK * NXP * 4);
  __bf16* dinb = (__bf16*)alloc((size_t)N_TOK * DI * 2);
  float*  dlt  = (float*) alloc((size_t)N_TOK * DI * 4);
  float*  ybuf = (float*) alloc((size_t)N_TOK * DI * 4);
  __bf16* ygb  = (__bf16*)alloc((size_t)N_TOK * DI * 2);

  auto cvtN = [&](const float* s, __bf16* d, size_t n) {
    cvt_bf16<<<(unsigned)((n + 255) / 256), 256, 0, stream>>>(s, d, (int)n);
  };
  cvtN(in_proj,  w1b, (size_t)DMD * 2 * DI);
  cvtN(x_proj,   w2b, (size_t)DI * NXP);
  cvtN(dt_w,     w3b, (size_t)DI * DI);
  cvtN(out_proj, w4b, (size_t)DI * DMD);

  // 1) LayerNorm -> bf16
  ln_kernel<<<N_TOK, 256, 0, stream>>>(x, norm_w, norm_b, xnb);

  // 2) xz = xn @ in_proj  (2048 x 4096, K=1024)
  {
    dim3 g((2 * DI) / 64, N_TOK / 128);
    gemm_bf16_wmma<<<g, 256, 0, stream>>>(xnb, w1b, xz, N_TOK, 2 * DI, DMD,
                                          0, nullptr, nullptr);
  }
  // 3) depthwise causal conv + SiLU
  conv_silu<<<(N_TOK * DI) / 256, 256, 0, stream>>>(xz, conv_w, conv_b, xbsf, xbsb);

  // 4) x_dbl = x_b @ x_proj (2048 x 2080, K=2048)
  {
    dim3 g((NXP + 63) / 64, N_TOK / 128);
    gemm_bf16_wmma<<<g, 256, 0, stream>>>(xbsb, w2b, xdbl, N_TOK, NXP, DI,
                                          0, nullptr, nullptr);
  }
  // 5) delta = softplus(delta_in @ dt_w + dt_b)
  din_cvt<<<(N_TOK * DI) / 256, 256, 0, stream>>>(xdbl, dinb);
  {
    dim3 g(DI / 64, N_TOK / 128);
    gemm_bf16_wmma<<<g, 256, 0, stream>>>(dinb, w3b, dlt, N_TOK, DI, DI,
                                          1, dt_b, nullptr);
  }
  // 6) selective scan
  scan_kernel<<<dim3(DI / 256, NB), 256, 0, stream>>>(dlt, xbsf, xdbl, A_log, ybuf);

  // 7) gate: (y + x_b*D) * SiLU(z) -> bf16
  gate_kernel<<<(N_TOK * DI) / 256, 256, 0, stream>>>(ybuf, xbsf, xz, D_param, ygb);

  // 8) out = yg @ out_proj + residual(x)
  {
    dim3 g(DMD / 64, N_TOK / 128);
    gemm_bf16_wmma<<<g, 256, 0, stream>>>(ygb, w4b, out, N_TOK, DMD, DI,
                                          2, nullptr, x);
  }
}
